// GraphUnetPool_74225624809765
// MI455X (gfx1250) — 8-bit matrix operand order repaired and checked by reading, compiled, not run on hardware
//
#include <hip/hip_runtime.h>
#include <math.h>

typedef __attribute__((ext_vector_type(8))) int v8i;
typedef __attribute__((ext_vector_type(4))) int v4i;
typedef __attribute__((ext_vector_type(2))) int v2i;

#define N_NODES 8192
#define DIM     256
#define KN      4096
#define NE      262144

__global__ void zero_u32(unsigned int* __restrict__ p, size_t n) {
    size_t i = (size_t)blockIdx.x * blockDim.x + threadIdx.x;
    size_t stride = (size_t)gridDim.x * blockDim.x;
    for (; i < n; i += stride) p[i] = 0u;
}

__global__ void fill_ones(float* __restrict__ p, int n) {
    int i = blockIdx.x * blockDim.x + threadIdx.x;
    int stride = gridDim.x * blockDim.x;
    for (; i < n; i += stride) p[i] = 1.0f;
}

__global__ void build_adj(const int* __restrict__ ei,
                          unsigned int* __restrict__ g32,
                          unsigned int* __restrict__ gT32) {
    int e = blockIdx.x * blockDim.x + threadIdx.x;
    if (e >= NE) return;
    unsigned int u = (unsigned int)ei[e];
    unsigned int v = (unsigned int)ei[NE + e];
    size_t a = (size_t)u * N_NODES + v;
    atomicAdd(g32 + (a >> 2), 1u << ((a & 3u) * 8u));
    size_t b = (size_t)v * N_NODES + u;
    atomicAdd(gT32 + (b >> 2), 1u << ((b & 3u) * 8u));
}

__global__ __launch_bounds__(256) void node_scores(const float* __restrict__ h,
                                                   const float* __restrict__ w,
                                                   const float* __restrict__ b,
                                                   float* __restrict__ scores) {
    int wave = threadIdx.x >> 5;
    int lane = threadIdx.x & 31;
    int n = blockIdx.x * 8 + wave;
    const float* hr = h + (size_t)n * DIM;
    float s = 0.f;
#pragma unroll
    for (int m = 0; m < 8; ++m)
        s += hr[m * 32 + lane] * w[m * 32 + lane];
#pragma unroll
    for (int off = 16; off > 0; off >>= 1)
        s += __shfl_xor(s, off, 32);
    if (lane == 0)
        scores[n] = 1.f / (1.f + __expf(-(s + b[0])));
}

__global__ __launch_bounds__(1024) void topk_sort(const float* __restrict__ scores,
                                                  const int* __restrict__ batch,
                                                  int* __restrict__ sIdx,
                                                  float* __restrict__ sVal,
                                                  float* __restrict__ outIdx,
                                                  float* __restrict__ outBatch) {
    __shared__ float key[N_NODES];
    __shared__ int   id [N_NODES];
    int tid = threadIdx.x;
    for (int i = tid; i < N_NODES; i += 1024) { key[i] = scores[i]; id[i] = i; }
    __syncthreads();
    for (int k = 2; k <= N_NODES; k <<= 1) {
        for (int j = k >> 1; j > 0; j >>= 1) {
            for (int i = tid; i < N_NODES; i += 1024) {
                int ixj = i ^ j;
                if (ixj > i) {
                    bool dir = ((i & k) == 0);
                    float ka = key[i], kb = key[ixj];
                    int   ia = id[i],  ib = id[ixj];
                    bool wrong = (kb > ka) || (kb == ka && ib < ia);
                    if (wrong == dir) {
                        key[i] = kb; key[ixj] = ka;
                        id[i]  = ib; id[ixj]  = ia;
                    }
                }
            }
            __syncthreads();
        }
    }
    for (int i = tid; i < N_NODES; i += 1024) { sIdx[i] = id[i]; sVal[i] = key[i]; }
    for (int i = tid; i < KN; i += 1024) {
        outIdx[i]   = (float)id[i];
        outBatch[i] = (float)batch[id[i]];
    }
}

__global__ __launch_bounds__(256) void gather_scale(const float* __restrict__ h,
                                                    const int* __restrict__ sIdx,
                                                    const float* __restrict__ sVal,
                                                    float* __restrict__ out) {
    int t = blockIdx.x * 256 + threadIdx.x;
    int row = t >> 6;
    int c4  = (t & 63) << 2;
    const float4 v = *(const float4*)(h + (size_t)sIdx[row] * DIM + c4);
    float s = sVal[row];
    float4 r;
    r.x = v.x * s; r.y = v.y * s; r.z = v.z * s; r.w = v.w * s;
    *(float4*)(out + (size_t)row * DIM + c4) = r;
}

#define BM   64
#define BN   256
#define KC   64
#define LDSS 80
#define A_BYTES (BM * LDSS)
#define B_BYTES (BN * LDSS)
#define STAGE   (A_BYTES + B_BYTES)
#define NCHUNK  (N_NODES / KC)

#define ASYNC_B128(ldsaddr, gaddr)                                       \
    asm volatile("global_load_async_to_lds_b128 %0, %1, off"             \
                 :: "v"(ldsaddr), "v"(gaddr) : "memory")

__global__ __launch_bounds__(256) void gemm_iu8(const unsigned char* __restrict__ g,
                                                const unsigned char* __restrict__ gT,
                                                const int* __restrict__ kidx,
                                                float* __restrict__ out) {
    __shared__ __align__(16) unsigned char smem[2][STAGE];
    __shared__ int rA[BM];
    __shared__ int rB[BN];

    const int tid    = threadIdx.x;
    const int lane   = tid & 31;
    const int wave   = tid >> 5;
    const int laneLo = lane & 15;
    const int laneHi = lane >> 4;
    const int waveM  = wave >> 2;
    const int waveN  = wave & 3;
    const int blockM = blockIdx.y;
    const int blockN = blockIdx.x;

    if (tid < BM) rA[tid] = kidx[blockM * BM + tid];
    rB[tid] = kidx[blockN * BN + tid];
    __syncthreads();

    const unsigned ldsBase = (unsigned)(uintptr_t)&smem[0][0];
    const int arow = tid >> 2, ac = (tid & 3) << 4;
    const unsigned long long gA = (unsigned long long)(uintptr_t)
        (g + (size_t)rA[arow] * N_NODES + ac);
    const unsigned ldsA = ldsBase + arow * LDSS + ac;

    unsigned long long gB[4];
    unsigned ldsB[4];
#pragma unroll
    for (int i = 0; i < 4; ++i) {
        int w2 = tid + 256 * i;
        int brow = w2 >> 2, bc = (w2 & 3) << 4;
        gB[i] = (unsigned long long)(uintptr_t)(gT + (size_t)rB[brow] * N_NODES + bc);
        ldsB[i] = ldsBase + A_BYTES + brow * LDSS + bc;
    }

    v8i acc[2][4];
#pragma unroll
    for (int mi = 0; mi < 2; ++mi)
#pragma unroll
        for (int ni = 0; ni < 4; ++ni)
#pragma unroll
            for (int r = 0; r < 8; ++r)
                acc[mi][ni][r] = 0;

    ASYNC_B128(ldsA, gA);
#pragma unroll
    for (int i = 0; i < 4; ++i) ASYNC_B128(ldsB[i], gB[i]);
    asm volatile("s_wait_asynccnt 0" ::: "memory");
    __syncthreads();

    for (int t = 0; t < NCHUNK; ++t) {
        const int kb  = t * KC;
        const int buf = t & 1;

        if (t + 1 < NCHUNK) {
            const unsigned bo = (buf ^ 1) ? (unsigned)STAGE : 0u;
            ASYNC_B128(ldsA + bo, gA + (unsigned long long)(kb + KC));
#pragma unroll
            for (int i = 0; i < 4; ++i)
                ASYNC_B128(ldsB[i] + bo, gB[i] + (unsigned long long)(kb + KC));
        }
        if (t + 2 < NCHUNK) {
            __builtin_prefetch((const void*)(uintptr_t)(gA + (unsigned long long)(kb + 2 * KC)), 0, 1);
            __builtin_prefetch((const void*)(uintptr_t)(gB[0] + (unsigned long long)(kb + 2 * KC)), 0, 1);
            __builtin_prefetch((const void*)(uintptr_t)(gB[2] + (unsigned long long)(kb + 2 * KC)), 0, 1);
        }

        const unsigned char* sA = &smem[buf][0];
        const unsigned char* sB = &smem[buf][A_BYTES];

        v8i aF[2];
#pragma unroll
        for (int mi = 0; mi < 2; ++mi) {
            const unsigned char* p = sA + (waveM * 32 + mi * 16 + laneLo) * LDSS + laneHi * 16;
            v2i q0 = *(const v2i*)(p);
            v2i q1 = *(const v2i*)(p + 8);
            v2i q2 = *(const v2i*)(p + 32);
            v2i q3 = *(const v2i*)(p + 40);
            aF[mi][0] = q0[0]; aF[mi][1] = q0[1];
            aF[mi][2] = q1[0]; aF[mi][3] = q1[1];
            aF[mi][4] = q2[0]; aF[mi][5] = q2[1];
            aF[mi][6] = q3[0]; aF[mi][7] = q3[1];
        }
        v8i bF[4];
#pragma unroll
        for (int ni = 0; ni < 4; ++ni) {
            const unsigned char* p = sB + (waveN * 64 + ni * 16 + laneLo) * LDSS + laneHi * 16;
            v4i r0 = *(const v4i*)(p);
            v4i r1 = *(const v4i*)(p + 32);
            bF[ni][0] = r0[0]; bF[ni][1] = r0[1]; bF[ni][2] = r0[2]; bF[ni][3] = r0[3];
            bF[ni][4] = r1[0]; bF[ni][5] = r1[1]; bF[ni][6] = r1[2]; bF[ni][7] = r1[3];
        }

#pragma unroll
        for (int mi = 0; mi < 2; ++mi)
#pragma unroll
            for (int ni = 0; ni < 4; ++ni)
                acc[mi][ni] = __builtin_amdgcn_wmma_i32_16x16x64_iu8(
                    false, aF[mi], false, bF[ni], acc[mi][ni], false, false);

        asm volatile("s_wait_asynccnt 0" ::: "memory");
        __syncthreads();
    }

#pragma unroll
    for (int mi = 0; mi < 2; ++mi)
#pragma unroll
        for (int ni = 0; ni < 4; ++ni) {
            int rowBase = blockM * BM + waveM * 32 + mi * 16 + laneHi * 8;
            int col     = blockN * BN + waveN * 64 + ni * 16 + laneLo;
#pragma unroll
            for (int r = 0; r < 8; ++r)
                out[(size_t)(rowBase + r) * KN + col] = (float)acc[mi][ni][r];
        }
}

extern "C" void kernel_launch(void* const* d_in, const int* in_sizes, int n_in,
                              void* d_out, int out_size, void* d_ws, size_t ws_size,
                              hipStream_t stream) {
    const float* h     = (const float*)d_in[0];
    const int*   ei    = (const int*)d_in[1];
    const int*   batch = (const int*)d_in[3];
    const float* pw    = (const float*)d_in[4];
    const float* pb    = (const float*)d_in[5];
    float* out = (float*)d_out;

    char* ws = (char*)d_ws;
    float* scores = (float*)ws;
    int*   sIdx   = (int*)(ws + 32768);
    float* sVal   = (float*)(ws + 65536);
    unsigned char* g  = (unsigned char*)(ws + (1 << 20));
    unsigned char* gT = g + (size_t)N_NODES * N_NODES;

    float* out_newh  = out;
    float* out_ung   = out_newh + (size_t)KN * DIM;
    float* out_ea    = out_ung + (size_t)KN * KN;
    float* out_batch = out_ea + NE;
    float* out_idx   = out_batch + KN;

    zero_u32<<<2048, 256, 0, stream>>>((unsigned int*)g,
                                       (size_t)2 * N_NODES * N_NODES / 4);
    build_adj<<<NE / 256, 256, 0, stream>>>(ei, (unsigned int*)g, (unsigned int*)gT);
    node_scores<<<N_NODES / 8, 256, 0, stream>>>(h, pw, pb, scores);
    topk_sort<<<1, 1024, 0, stream>>>(scores, batch, sIdx, sVal, out_idx, out_batch);
    gather_scale<<<(KN * (DIM / 4)) / 256, 256, 0, stream>>>(h, sIdx, sVal, out_newh);
    fill_ones<<<256, 256, 0, stream>>>(out_ea, NE);
    dim3 ggrid(KN / BN, KN / BM);
    gemm_iu8<<<ggrid, 256, 0, stream>>>(g, gT, sIdx, out_ung);
}
